// Attention_50766513438786
// MI455X (gfx1250) — compile-verified, matrix operand order checked by reading, not run on hardware
//
#include <hip/hip_runtime.h>
#include <hip/hip_bf16.h>

typedef __attribute__((ext_vector_type(16))) _Float16 v16h;
typedef __attribute__((ext_vector_type(8)))  float    v8f;
typedef __attribute__((ext_vector_type(4)))  int      v4i;
typedef __attribute__((ext_vector_type(8)))  int      v8i;
typedef __attribute__((ext_vector_type(4)))  unsigned v4u;

union V16H { v16h v; unsigned u[8]; };

// K index of the j=0 element of pair-register v (0..7) for 16-bit A/B WMMA frags.
// (ISA 7.12.2): VGPR0..3 lanes0-15 K=0..7 pairs, lanes16-31 K=8..15;
// VGPR4..7 lanes0-15 K=16..23, lanes16-31 K=24..31.
__device__ __forceinline__ int frag_k(int v, int hi) {
    int base = (v < 4) ? (v * 2) : (16 + (v - 4) * 2);
    return base + (hi ? 8 : 0);
}

// ---------------------------------------------------------------------------
// CDNA5 data movers.
//  Tier 1: TDM (TENSOR_LOAD_TO_LDS, TENSORcnt)  - one descriptor per tile.
//  Tier 2: GLOBAL_LOAD_ASYNC_TO_LDS_B128 (ASYNCcnt) - per-lane async copy.
//  Tier 3: synchronous b128 copies.
// ---------------------------------------------------------------------------
#if defined(__AMDGCN__) && \
    __has_builtin(__builtin_amdgcn_tensor_load_to_lds) && \
    __has_builtin(__builtin_amdgcn_s_wait_tensorcnt)
#define USE_TDM 1
#else
#define USE_TDM 0
#endif

#if defined(__AMDGCN__) && \
    __has_builtin(__builtin_amdgcn_global_load_async_to_lds_b128) && \
    __has_builtin(__builtin_amdgcn_s_wait_asynccnt)
#define USE_ASYNC 1
#else
#define USE_ASYNC 0
#endif

typedef __attribute__((address_space(1))) v4i gv4i;
typedef __attribute__((address_space(3))) v4i lv4i;
typedef __attribute__((address_space(3))) const void lc_void;

__device__ __forceinline__ unsigned lds_addr_of(const void* p) {
    return (unsigned)(unsigned long long)(lc_void*)p;
}

__device__ __forceinline__ void async_copy16(const _Float16* g, _Float16* l) {
#if USE_ASYNC
    __builtin_amdgcn_global_load_async_to_lds_b128(
        (gv4i*)(v4i*)(void*)g, (lv4i*)(v4i*)(void*)l, 0, 0);
#else
    *(float4*)l = *(const float4*)g;
#endif
}

__device__ __forceinline__ void wait_async_all() {
#if USE_ASYNC
    __builtin_amdgcn_s_wait_asynccnt(0);
#endif
}

// 2D TDM tile load: tile_d1 rows x tile_d0 elems (2-byte), row stride stride0,
// packed densely row-major into LDS at lds_byte_addr. D# per ISA ch.8.
// This toolchain's builtin is the 6-arg form:
//   (uint32x4 g0, int32x8 g1, int32x4 g2, int32x4 g3, int32x8 g4, i32 cpol)
__device__ __forceinline__ void tdm_load_2d(const void* g, unsigned lds_byte_addr,
                                            unsigned tile_d0, unsigned tile_d1,
                                            unsigned tensor_d0, unsigned tensor_d1,
                                            unsigned stride0) {
#if USE_TDM
    unsigned long long ga = (unsigned long long)g;
    v4u g0;
    g0.x = 1u;                                   // count=1 valid descriptor
    g0.y = lds_byte_addr;                        // lds_addr
    g0.z = (unsigned)ga;                         // global_addr[31:0]
    g0.w = (unsigned)(ga >> 32) | (2u << 30);    // global_addr[56:32] | type=2
    v8i g1;
    g1[0] = (int)(1u << 16);                     // data_size=1 (2 bytes), mask=0
    g1[1] = (int)((tensor_d0 & 0xffffu) << 16);              // tensor_dim0 lo
    g1[2] = (int)((tensor_d0 >> 16) | ((tensor_d1 & 0xffffu) << 16));
    g1[3] = (int)((tensor_d1 >> 16) | (tile_d0 << 16));      // tile_dim0
    g1[4] = (int)(tile_d1 & 0xffffu);                        // tile_dim1 (dim2=0)
    g1[5] = (int)stride0;                                    // dim0_stride[31:0]
    g1[6] = 0;
    g1[7] = 0;
    v4i zz4 = {0, 0, 0, 0};
    v8i zz8 = {0, 0, 0, 0, 0, 0, 0, 0};
    __builtin_amdgcn_tensor_load_to_lds(g0, g1, zz4, zz4, zz8, 0);
#else
    (void)g; (void)lds_byte_addr; (void)tile_d0; (void)tile_d1;
    (void)tensor_d0; (void)tensor_d1; (void)stride0;
#endif
}

__device__ __forceinline__ void wait_tensor_all() {
#if USE_TDM
    __builtin_amdgcn_s_wait_tensorcnt(0);
#endif
}

// ---------------------------------------------------------------------------
// f32 -> f16 conversion, 4 elems/thread, sizes guaranteed multiple of 1024
// ---------------------------------------------------------------------------
__global__ void cvt_f32_f16_kernel(const float* __restrict__ src,
                                   _Float16* __restrict__ dst, size_t n) {
    size_t i = ((size_t)blockIdx.x * blockDim.x + threadIdx.x) * 4;
    if (i + 3 < n) {
        float4 f = *(const float4*)(src + i);
        union { _Float16 h[4]; float2 f2; } u;
        u.h[0] = (_Float16)f.x; u.h[1] = (_Float16)f.y;
        u.h[2] = (_Float16)f.z; u.h[3] = (_Float16)f.w;
        *(float2*)(dst + i) = u.f2;
    }
}

// ---------------------------------------------------------------------------
// GEMM: out[M,Nc] = A[M,K](f16) * W[Nc,K]^T(f16) + bias, f32 out.
// 256 threads (8 waves), block tile 128x128, BK=32 (one WMMA K-step),
// double-buffered TDM (or async) LDS staging, each wave 32x64 (2x4 WMMA).
// ---------------------------------------------------------------------------
#define GBM 128
#define GBN 128
#define GBK 32

__global__ __launch_bounds__(256) void gemm_f16_kernel(
    const _Float16* __restrict__ A, const _Float16* __restrict__ W,
    const float* __restrict__ bias, float* __restrict__ out,
    int M, int Nc, int K)
{
    __shared__ __align__(16) _Float16 As[2][GBM * GBK];
    __shared__ __align__(16) _Float16 Bs[2][GBN * GBK];

    const int tid  = threadIdx.x;
    const int lane = tid & 31;
    const int wave = tid >> 5;
    const int wm = wave >> 1;      // 0..3  (32-row strip)
    const int wn = wave & 1;       // 0..1  (64-col strip)
    const int hi = lane >> 4;
    const int ln = lane & 15;
    const int bm = blockIdx.y * GBM;
    const int bn = blockIdx.x * GBN;

#if !USE_TDM
    const int srow = tid >> 1;            // 0..127
    const int scol = (tid & 1) * 16;      // 0 or 16
#endif

    auto stage = [&](int buf, int k0) {
#if USE_TDM
        if (wave == 0) {
            tdm_load_2d(A + (size_t)bm * K + k0, lds_addr_of(&As[buf][0]),
                        GBK, GBM, (unsigned)K, GBM, (unsigned)K);
            tdm_load_2d(W + (size_t)bn * K + k0, lds_addr_of(&Bs[buf][0]),
                        GBK, GBN, (unsigned)K, GBN, (unsigned)K);
        }
#else
        const _Float16* ga = A + (size_t)(bm + srow) * K + k0 + scol;
        _Float16* la = &As[buf][srow * GBK + scol];
        async_copy16(ga, la);
        async_copy16(ga + 8, la + 8);
        const _Float16* gb = W + (size_t)(bn + srow) * K + k0 + scol;
        _Float16* lb = &Bs[buf][srow * GBK + scol];
        async_copy16(gb, lb);
        async_copy16(gb + 8, lb + 8);
#endif
    };

    auto stage_wait = [&]() {
#if USE_TDM
        if (wave == 0) wait_tensor_all();
#else
        wait_async_all();
#endif
    };

    v8f acc[2][4] = {};
    const int steps = K / GBK;
    int buf = 0;
    stage(0, 0);

    for (int i = 0; i < steps; ++i) {
        stage_wait();
        __syncthreads();                      // tile i visible to all waves
        if (i + 1 < steps) stage(buf ^ 1, (i + 1) * GBK);

        V16H af[2], bf[4];
#pragma unroll
        for (int t = 0; t < 2; ++t) {
            int m = wm * 32 + t * 16 + ln;
#pragma unroll
            for (int v = 0; v < 8; ++v)
                af[t].u[v] = *(const unsigned*)(&As[buf][m * GBK + frag_k(v, hi)]);
        }
#pragma unroll
        for (int t = 0; t < 4; ++t) {
            int n = wn * 64 + t * 16 + ln;
#pragma unroll
            for (int v = 0; v < 8; ++v)
                bf[t].u[v] = *(const unsigned*)(&Bs[buf][n * GBK + frag_k(v, hi)]);
        }
#pragma unroll
        for (int tm = 0; tm < 2; ++tm)
#pragma unroll
            for (int tn = 0; tn < 4; ++tn)
                acc[tm][tn] = __builtin_amdgcn_wmma_f32_16x16x32_f16(
                    false, af[tm].v, false, bf[tn].v,
                    (short)0, acc[tm][tn], false, false);
        buf ^= 1;
    }

#pragma unroll
    for (int tm = 0; tm < 2; ++tm)
#pragma unroll
        for (int tn = 0; tn < 4; ++tn) {
            int col = bn + wn * 64 + tn * 16 + ln;
            float b = bias ? bias[col] : 0.f;
#pragma unroll
            for (int r = 0; r < 8; ++r) {
                int row = bm + wm * 32 + tm * 16 + r + hi * 8;
                out[(size_t)row * Nc + col] = acc[tm][tn][r] + b;
            }
        }
}

// ---------------------------------------------------------------------------
// QKV post-process: RMSNorm(q,k) + 3D RoPE, write q/k as f16 [B,H,N,D],
// v transposed as f16 [B,H,D,N]. One block per (b*N+n, h), 128 threads (d).
// ---------------------------------------------------------------------------
__global__ __launch_bounds__(128) void qkv_post_kernel(
    const float* __restrict__ qkv,
    const float* __restrict__ qw, const float* __restrict__ kw,
    _Float16* __restrict__ qh, _Float16* __restrict__ kh,
    _Float16* __restrict__ vt,
    const int* __restrict__ tszp, const int* __restrict__ hszp,
    const int* __restrict__ wszp)
{
    const int Hn = 24, D = 128, N = 2048, C = 3072, C3 = 9216;
    const int bn = blockIdx.x;          // b*N + n
    const int h  = blockIdx.y;
    const int d  = threadIdx.x;
    const int b  = bn / N, n = bn % N;

    const float* row = qkv + (size_t)bn * C3;
    float q = row[h * D + d];
    float k = row[C + h * D + d];
    float v = row[2 * C + h * D + d];

    __shared__ float red[8];
    const int lane = d & 31, wv = d >> 5;
    float sq = q * q, sk = k * k;
#pragma unroll
    for (int m = 16; m; m >>= 1) {
        sq += __shfl_xor(sq, m, 32);
        sk += __shfl_xor(sk, m, 32);
    }
    if (lane == 0) { red[wv] = sq; red[4 + wv] = sk; }
    __syncthreads();
    float msq = (red[0] + red[1] + red[2] + red[3]) * (1.f / 128.f);
    float msk = (red[4] + red[5] + red[6] + red[7]) * (1.f / 128.f);
    q = q * rsqrtf(msq + 1e-6f) * qw[d];
    k = k * rsqrtf(msk + 1e-6f) * kw[d];

    // 3D RoPE: dim_t=44 (22 pairs), dim_h=dim_w=42 (21 pairs each)
    const int tsz = *tszp, hsz = *hszp, wsz = *wszp;
    (void)tsz;
    int t  = n / (hsz * wsz);
    int rm = n % (hsz * wsz);
    int hh = rm / wsz, ww = rm % wsz;
    int p = d >> 1;
    float f;
    if (p < 22)       f = (float)t  * __powf(10000.f, -(float)(2 * p)        / 44.f);
    else if (p < 43)  f = (float)hh * __powf(10000.f, -(float)(2 * (p - 22)) / 42.f);
    else              f = (float)ww * __powf(10000.f, -(float)(2 * (p - 43)) / 42.f);
    float cs = __cosf(f), sn = __sinf(f);
    float qp = __shfl_xor(q, 1, 32);
    float kp = __shfl_xor(k, 1, 32);
    float qr, kr;
    if (d & 1) { qr = q * cs + qp * sn; kr = k * cs + kp * sn; }
    else       { qr = q * cs - qp * sn; kr = k * cs - kp * sn; }

    size_t base = ((size_t)(b * Hn + h) * N + n) * D + d;
    qh[base] = (_Float16)qr;
    kh[base] = (_Float16)kr;
    vt[((size_t)(b * Hn + h) * D + d) * N + n] = (_Float16)v;
}

// ---------------------------------------------------------------------------
// Flash attention: 256 threads (8 waves), 128 queries/block (16 per wave).
// Double-buffered K/V^T tiles (32 keys) staged by TDM (or async); per 32-key
// step each wave does 8 WMMAs (QK^T) + 8 WMMAs (P*V). Output f16 [B,N,H*D].
// ---------------------------------------------------------------------------
__global__ __launch_bounds__(256) void attn_kernel(
    const _Float16* __restrict__ qh, const _Float16* __restrict__ kh,
    const _Float16* __restrict__ vt, _Float16* __restrict__ outh)
{
    const int Hn = 24, D = 128, N = 2048, C = 3072;
    const float scale = 0.08838834764831845f;   // 128^-0.5

    __shared__ __align__(16) _Float16 Ks[2][32 * 128];   // [key][d]
    __shared__ __align__(16) _Float16 Vs[2][128 * 32];   // [d][key]
    __shared__ __align__(16) _Float16 Ps[8][16 * 32];    // per-wave P tile

    const int tid  = threadIdx.x;
    const int lane = tid & 31;
    const int wave = tid >> 5;
    const int hi = lane >> 4, ln = lane & 15;
    const int bh = blockIdx.y;                 // b*Hn + h
    const int qbase = blockIdx.x * 128 + wave * 16;

#if !USE_TDM
    const int krow = tid >> 3, kcol = (tid & 7) * 16;   // K: [32][128]
    const int vrow = tid >> 1, vcol = (tid & 1) * 16;   // V^T: [128][32]
#endif

    auto stage = [&](int buf, int kt) {
#if USE_TDM
        if (wave == 0) {
            tdm_load_2d(kh + ((size_t)bh * N + kt) * D, lds_addr_of(&Ks[buf][0]),
                        D, 32, D, 32, D);
            tdm_load_2d(vt + (size_t)bh * D * N + kt, lds_addr_of(&Vs[buf][0]),
                        32, D, 32, D, N);
        }
#else
        const _Float16* kg = kh + ((size_t)bh * N + kt + krow) * D + kcol;
        _Float16* ks = &Ks[buf][krow * 128 + kcol];
        async_copy16(kg, ks);
        async_copy16(kg + 8, ks + 8);
        const _Float16* vg = vt + ((size_t)bh * D + vrow) * N + kt + vcol;
        _Float16* vs = &Vs[buf][vrow * 32 + vcol];
        async_copy16(vg, vs);
        async_copy16(vg + 8, vs + 8);
#endif
    };

    auto stage_wait = [&]() {
#if USE_TDM
        if (wave == 0) wait_tensor_all();
#else
        wait_async_all();
#endif
    };

    // Q fragments resident in registers: 16 rows x D=128 -> 4 K-chunks
    const _Float16* qrow = qh + (size_t)bh * N * D;
    V16H qf[4];
#pragma unroll
    for (int c = 0; c < 4; ++c)
#pragma unroll
        for (int v = 0; v < 8; ++v) {
            int kk = c * 32 + frag_k(v, hi);
            qf[c].u[v] = *(const unsigned*)(qrow + (size_t)(qbase + ln) * D + kk);
        }

    v8f O[8] = {};
    float mrow[8], srow[8];
#pragma unroll
    for (int r = 0; r < 8; ++r) { mrow[r] = -1e30f; srow[r] = 0.f; }

    int buf = 0;
    stage(0, 0);

    for (int kt = 0; kt < N; kt += 32) {
        stage_wait();
        __syncthreads();
        if (kt + 32 < N) stage(buf ^ 1, kt + 32);

        // S = Q * K^T for two 16-key subtiles
        v8f sa[2] = {};
#pragma unroll
        for (int sub = 0; sub < 2; ++sub) {
#pragma unroll
            for (int c = 0; c < 4; ++c) {
                V16H bf;
                int nloc = sub * 16 + ln;
#pragma unroll
                for (int v = 0; v < 8; ++v) {
                    int kk = c * 32 + frag_k(v, hi);
                    bf.u[v] = *(const unsigned*)(&Ks[buf][nloc * 128 + kk]);
                }
                sa[sub] = __builtin_amdgcn_wmma_f32_16x16x32_f16(
                    false, qf[c].v, false, bf.v, (short)0, sa[sub], false, false);
            }
        }

        // joint online softmax over the 32 keys
        float p0[8], p1[8];
#pragma unroll
        for (int r = 0; r < 8; ++r) {
            float s0 = sa[0][r] * scale, s1 = sa[1][r] * scale;
            float mx = fmaxf(s0, s1);
#pragma unroll
            for (int m = 1; m < 16; m <<= 1) mx = fmaxf(mx, __shfl_xor(mx, m, 32));
            float mn = fmaxf(mrow[r], mx);
            float alpha = __expf(mrow[r] - mn);
            float e0 = __expf(s0 - mn), e1 = __expf(s1 - mn);
            float rs = e0 + e1;
#pragma unroll
            for (int m = 1; m < 16; m <<= 1) rs += __shfl_xor(rs, m, 32);
            srow[r] = srow[r] * alpha + rs;
            mrow[r] = mn;
#pragma unroll
            for (int c = 0; c < 8; ++c) O[c][r] *= alpha;
            p0[r] = e0; p1[r] = e1;
        }

        // C-layout -> A-layout through per-wave LDS tile
#pragma unroll
        for (int r = 0; r < 8; ++r) {
            int m = r + hi * 8;
            Ps[wave][m * 32 + ln]      = (_Float16)p0[r];
            Ps[wave][m * 32 + 16 + ln] = (_Float16)p1[r];
        }

        V16H pf;
#pragma unroll
        for (int v = 0; v < 8; ++v)
            pf.u[v] = *(const unsigned*)(&Ps[wave][ln * 32 + frag_k(v, hi)]);

#pragma unroll
        for (int c = 0; c < 8; ++c) {
            V16H vf;
            int dcol = c * 16 + ln;
#pragma unroll
            for (int v = 0; v < 8; ++v)
                vf.u[v] = *(const unsigned*)(&Vs[buf][dcol * 32 + frag_k(v, hi)]);
            O[c] = __builtin_amdgcn_wmma_f32_16x16x32_f16(
                false, pf.v, false, vf.v, (short)0, O[c], false, false);
        }
        buf ^= 1;
    }

    // normalize and write out as f16 [B, N, H*D]
    const int b = bh / Hn, h = bh % Hn;
#pragma unroll
    for (int r = 0; r < 8; ++r) {
        float inv = 1.f / srow[r];
        int qg = qbase + r + hi * 8;
        size_t rowoff = ((size_t)b * N + qg) * C + (size_t)h * D;
#pragma unroll
        for (int c = 0; c < 8; ++c)
            outh[rowoff + c * 16 + ln] = (_Float16)(O[c][r] * inv);
    }
}

// ---------------------------------------------------------------------------
extern "C" void kernel_launch(void* const* d_in, const int* in_sizes, int n_in,
                              void* d_out, int out_size, void* d_ws, size_t ws_size,
                              hipStream_t stream) {
    const float* x        = (const float*)d_in[0];
    const float* qkv_w    = (const float*)d_in[1];
    const float* qkv_b    = (const float*)d_in[2];
    const float* q_norm_w = (const float*)d_in[3];
    const float* k_norm_w = (const float*)d_in[4];
    const float* proj_w   = (const float*)d_in[5];
    const float* proj_b   = (const float*)d_in[6];
    const int*   tsz      = (const int*)d_in[7];
    const int*   hsz      = (const int*)d_in[8];
    const int*   wsz      = (const int*)d_in[9];
    float* out = (float*)d_out;

    const size_t Bv = 2, Nv = 2048, Cv = 3072, C3 = 9216, Hn = 24, Dv = 128;
    const size_t BN = Bv * Nv;               // 4096
    const size_t HND = Bv * Hn * Nv * Dv;    // 12,582,912

    char* ws = (char*)d_ws;
    _Float16* x_h    = (_Float16*)ws; ws += BN * Cv * sizeof(_Float16);
    _Float16* wq_h   = (_Float16*)ws; ws += C3 * Cv * sizeof(_Float16);
    _Float16* wp_h   = (_Float16*)ws; ws += Cv * Cv * sizeof(_Float16);
    float*    qkv_f  = (float*)ws;    ws += BN * C3 * sizeof(float);
    _Float16* q_h    = (_Float16*)ws; ws += HND * sizeof(_Float16);
    _Float16* k_h    = (_Float16*)ws; ws += HND * sizeof(_Float16);
    _Float16* vt_h   = (_Float16*)ws; ws += HND * sizeof(_Float16);
    _Float16* attn_h = (_Float16*)ws; ws += BN * Cv * sizeof(_Float16);

    cvt_f32_f16_kernel<<<(unsigned)(BN * Cv / 1024), 256, 0, stream>>>(x, x_h, BN * Cv);
    cvt_f32_f16_kernel<<<(unsigned)(C3 * Cv / 1024), 256, 0, stream>>>(qkv_w, wq_h, C3 * Cv);
    cvt_f32_f16_kernel<<<(unsigned)(Cv * Cv / 1024), 256, 0, stream>>>(proj_w, wp_h, Cv * Cv);

    // QKV projection: [4096,3072] x [9216,3072]^T -> [4096,9216]
    gemm_f16_kernel<<<dim3((unsigned)(C3 / GBN), (unsigned)(BN / GBM)), 256, 0, stream>>>(
        x_h, wq_h, qkv_b, qkv_f, (int)BN, (int)C3, (int)Cv);

    // RMSNorm + RoPE + layout
    qkv_post_kernel<<<dim3((unsigned)BN, (unsigned)Hn), 128, 0, stream>>>(
        qkv_f, q_norm_w, k_norm_w, q_h, k_h, vt_h, tsz, hsz, wsz);

    // Flash attention (128 queries per block)
    attn_kernel<<<dim3((unsigned)(Nv / 128), (unsigned)(Bv * Hn)), 256, 0, stream>>>(
        q_h, k_h, vt_h, attn_h);

    // Output projection: [4096,3072] x [3072,3072]^T -> d_out
    gemm_f16_kernel<<<dim3((unsigned)(Cv / GBN), (unsigned)(BN / GBM)), 256, 0, stream>>>(
        attn_h, wp_h, proj_b, out, (int)BN, (int)Cv, (int)Cv);
}
